// Attention_layer_1056561955134
// MI455X (gfx1250) — matrix operand order repaired and checked by reading, not compiled, not run on hardware
//
#include <hip/hip_runtime.h>
#include <hip/hip_bf16.h>

typedef __bf16 bf16_t;
typedef __attribute__((ext_vector_type(16))) __bf16 v16bf;
typedef __attribute__((ext_vector_type(4)))  __bf16 v4bf;
typedef __attribute__((ext_vector_type(2)))  __bf16 v2bf;
typedef __attribute__((ext_vector_type(8)))  float  v8f;
typedef __attribute__((ext_vector_type(4)))  float  v4f;

#define BATCH    16
#define TDLEN    1024
#define TELEN    2048
#define DDIM     1024
#define BM       32           // decoder rows per workgroup
#define BN       32           // encoder rows per tile iteration
#define NTHREADS 256
#define NWAVES   8
#define DCHUNK   (DDIM / NWAVES)   // 128 output columns per wave

// ---- dynamic LDS layout (bytes) ----
#define QS_OFF  0                                   // [BM][DDIM] bf16 (natural) = 65536
#define ES_OFF  (QS_OFF + BM * DDIM * 2)            // [BN][DDIM] bf16 (natural)    = 65536
#define ET_OFF  (ES_OFF + BN * DDIM * 2)            // [DDIM][BN] bf16 (natural)    = 65536
#define SS_OFF  (ET_OFF + DDIM * BN * 2)            // [2][BM][BN] f32 (split-K)    = 8192
#define PS_OFF  (SS_OFF + 2 * BM * BN * 4)          // [BM][BN] bf16 (natural)   = 2048
#define RED_OFF (PS_OFF + BM * BN * 2)              // [BM][8] f32                  = 1024
#define M_OFF   (RED_OFF + BM * 8 * 4)
#define L_OFF   (M_OFF + BM * 4)
#define A_OFF   (L_OFF + BM * 4)
#define SMEM_BYTES (A_OFF + BM * 4)                 // = 208256

__device__ __forceinline__ v8f wmma_bf16(v16bf a, v16bf b, v8f c) {
    return __builtin_amdgcn_wmma_f32_16x16x32_bf16(
        false, a, false, b, (short)0, c, false, false);
}

// packed f32x4 -> bf16x4 conversion (uses v_cvt_pk_bf16_f32 when available)
__device__ __forceinline__ v4bf cvt4_bf16(v4f v) {
#if __has_builtin(__builtin_amdgcn_cvt_pk_bf16_f32)
    v2bf lo = __builtin_amdgcn_cvt_pk_bf16_f32(v.x, v.y);
    v2bf hi = __builtin_amdgcn_cvt_pk_bf16_f32(v.z, v.w);
    return (v4bf){ lo[0], lo[1], hi[0], hi[1] };
#else
    return (v4bf){ (bf16_t)v.x, (bf16_t)v.y, (bf16_t)v.z, (bf16_t)v.w };
#endif
}

// A-side group-of-8 permutation within a K=32 chunk: [g0, g2, g1, g3]
__device__ __forceinline__ int perm_slot(int g) {
    return (g == 1) ? 2 : (g == 2 ? 1 : g);
}

__global__ void __launch_bounds__(NTHREADS)
attn_flash_kernel(const float* __restrict__ enc,   // [B, Te, D]
                  const float* __restrict__ dec,   // [B, Td, D]
                  float* __restrict__ out)         // [B, Td, 2D]
{
    extern __shared__ char smem[];
    bf16_t* Qs   = (bf16_t*)(smem + QS_OFF);
    bf16_t* Es   = (bf16_t*)(smem + ES_OFF);
    bf16_t* ET   = (bf16_t*)(smem + ET_OFF);
    float*  Ss   = (float*) (smem + SS_OFF);
    bf16_t* Ps   = (bf16_t*)(smem + PS_OFF);
    float*  red  = (float*) (smem + RED_OFF);
    float*  Mrow = (float*) (smem + M_OFF);
    float*  Lrow = (float*) (smem + L_OFF);
    float*  Arow = (float*) (smem + A_OFF);

    const int b    = blockIdx.x;
    const int row0 = blockIdx.y * BM;
    const int tid  = threadIdx.x;
    const int lane = tid & 31;
    const int wid  = tid >> 5;            // 0..7
    const int hi   = (lane >> 4) & 1;
    const int l16  = lane & 15;

    // ---------------- stage Q tile (f32 -> bf16, natural order) ----------------
    const float* qglob = dec + ((size_t)(b * TDLEN + row0)) * DDIM;
    #pragma unroll
    for (int i = 0; i < (BM * DDIM) / (NTHREADS * 4); ++i) {   // 32 iters
        int idx = (tid + i * NTHREADS) * 4;
        int r = idx >> 10;
        int c = idx & (DDIM - 1);
        v4f q = *(const v4f*)(qglob + idx);
        int dst = r * DDIM + c;
        *(v4bf*)(Qs + dst) = cvt4_bf16(q);
    }
    if (tid < BM) { Mrow[tid] = -1e30f; Lrow[tid] = 0.0f; }

    // output accumulators: wave owns D columns [wid*128, wid*128+128)
    v8f O[2][8];
    #pragma unroll
    for (int mt = 0; mt < 2; ++mt)
        #pragma unroll
        for (int nt = 0; nt < 8; ++nt)
            O[mt][nt] = (v8f){0.f,0.f,0.f,0.f,0.f,0.f,0.f,0.f};

    // S sub-tile assignment: 4 sub-tiles x 2 K-halves across 8 waves
    const int st   = wid & 3;
    const int mt_s = st >> 1;
    const int nt_s = st & 1;
    const int kh   = wid >> 2;            // K-half: [kh*512, kh*512+512)

    // ================ main loop over encoder tiles ================
    for (int n0 = 0; n0 < TELEN; n0 += BN) {
        __syncthreads();   // Es/ET/Ps/Ss free from previous iteration

        // ---- stage E tile: 4x4 register-transposed blocks ----
        const float* eglob = enc + ((size_t)(b * TELEN + n0)) * DDIM;
        #pragma unroll
        for (int i = 0; i < BN / 4; ++i) {            // 8 iters: rows i*4..i*4+3
            const int r0 = i * 4;
            const int c0 = tid * 4;                   // cols c0..c0+3
            v4f e0 = *(const v4f*)(eglob + (r0 + 0) * DDIM + c0);
            v4f e1 = *(const v4f*)(eglob + (r0 + 1) * DDIM + c0);
            v4f e2 = *(const v4f*)(eglob + (r0 + 2) * DDIM + c0);
            v4f e3 = *(const v4f*)(eglob + (r0 + 3) * DDIM + c0);
            v4bf b0 = cvt4_bf16(e0);
            v4bf b1 = cvt4_bf16(e1);
            v4bf b2 = cvt4_bf16(e2);
            v4bf b3 = cvt4_bf16(e3);
            // row-major copy (natural order) for GEMM1 B-fragments
            *(v4bf*)(Es + (r0 + 0) * DDIM + c0) = b0;
            *(v4bf*)(Es + (r0 + 1) * DDIM + c0) = b1;
            *(v4bf*)(Es + (r0 + 2) * DDIM + c0) = b2;
            *(v4bf*)(Es + (r0 + 3) * DDIM + c0) = b3;
            // transposed copy (d-major) for GEMM2 B-fragments
            #pragma unroll
            for (int l = 0; l < 4; ++l) {
                v4bf t = { b0[l], b1[l], b2[l], b3[l] };
                *(v4bf*)(ET + (c0 + l) * BN + r0) = t;
            }
        }
        // prefetch next E tile into cache while we compute on this one
        if (n0 + BN < TELEN) {
            const float* nxt = enc + ((size_t)(b * TELEN + n0 + BN)) * DDIM;
            #pragma unroll
            for (int j = 0; j < 4; ++j)
                __builtin_prefetch(nxt + (size_t)(tid + j * NTHREADS) * 32, 0, 0);
        }
        __syncthreads();   // Es/ET ready

        // ---- GEMM1 (split-K): S = Q * E^T ----
        {
            v8f sacc = (v8f){0.f,0.f,0.f,0.f,0.f,0.f,0.f,0.f};
            const bf16_t* qrow = Qs + (mt_s * 16 + l16) * DDIM;   // natural
            const bf16_t* erow = Es + (nt_s * 16 + l16) * DDIM;   // natural
            const int dlo = kh * (DDIM / 2);
            for (int d0 = dlo; d0 < dlo + DDIM / 2; d0 += 32) {   // 16 iters
                v16bf afrag = *(const v16bf*)(qrow + d0 + 16 * hi);
                v16bf bfrag = *(const v16bf*)(erow + d0 + 16 * hi);
                sacc = wmma_bf16(afrag, bfrag, sacc);
            }
            float* sdst = Ss + kh * (BM * BN);
            #pragma unroll
            for (int v = 0; v < 8; ++v)
                sdst[(mt_s * 16 + v + 8 * hi) * BN + nt_s * 16 + l16] = sacc[v];
        }
        __syncthreads();   // both S halves ready

        // ---- online softmax: combine split-K halves + row max ----
        const int srow_i = tid >> 3;          // 0..31
        const int cg     = tid & 7;           // 8 threads/row, 4 cols each
        float* s0 = Ss + srow_i * BN + cg * 4;
        {
            const float* s1 = s0 + BM * BN;
            float pm = -1e30f;
            #pragma unroll
            for (int c = 0; c < 4; ++c) {
                float s = s0[c] + s1[c];
                s0[c] = s;                    // combined S kept in half 0
                pm = fmaxf(pm, s);
            }
            red[srow_i * 8 + cg] = pm;
        }
        __syncthreads();
        if (tid < BM) {
            float m_old = Mrow[tid];
            float mn = m_old;
            #pragma unroll
            for (int g = 0; g < 8; ++g) mn = fmaxf(mn, red[tid * 8 + g]);
            Arow[tid] = __expf(m_old - mn);
            Mrow[tid] = mn;
        }
        __syncthreads();   // Mrow/Arow ready

        // ---- P = exp(S - m) (natural-order store), partial row sums ----
        {
            float m_r = Mrow[srow_i];
            float p0 = __expf(s0[0] - m_r);
            float p1 = __expf(s0[1] - m_r);
            float p2 = __expf(s0[2] - m_r);
            float p3 = __expf(s0[3] - m_r);
            v4bf pp = cvt4_bf16((v4f){p0, p1, p2, p3});
            int slot = cg >> 1;
            *(v4bf*)(Ps + srow_i * BN + slot * 8 + (cg & 1) * 4) = pp;
            red[srow_i * 8 + cg] = p0 + p1 + p2 + p3;
        }
        __syncthreads();   // Ps/red ready
        if (tid < BM) {
            float s = 0.0f;
            #pragma unroll
            for (int g = 0; g < 8; ++g) s += red[tid * 8 + g];
            Lrow[tid] = Lrow[tid] * Arow[tid] + s;
        }

        // ---- rescale O accumulators by alpha[row] ----
        #pragma unroll
        for (int mt = 0; mt < 2; ++mt)
            #pragma unroll
            for (int v = 0; v < 8; ++v) {
                float al = Arow[mt * 16 + v + 8 * hi];
                #pragma unroll
                for (int nt = 0; nt < 8; ++nt) O[mt][nt][v] *= al;
            }

        // ---- GEMM2: O(chunk) += P * E[:, chunk]  (K = BN = 32) ----
        {
            const int dbase = wid * DCHUNK;
            v16bf pa[2];
            #pragma unroll
            for (int mt = 0; mt < 2; ++mt)
                pa[mt] = *(const v16bf*)(Ps + (mt * 16 + l16) * BN + 16 * hi);
            #pragma unroll
            for (int nt = 0; nt < 8; ++nt) {
                const int dcol = dbase + nt * 16 + l16;
                v16bf bfrag = *(const v16bf*)(ET + dcol * BN + 16 * hi);
                O[0][nt] = wmma_bf16(pa[0], bfrag, O[0][nt]);
                O[1][nt] = wmma_bf16(pa[1], bfrag, O[1][nt]);
            }
        }
    }
    __syncthreads();   // Lrow final

    // ------- epilogue: normalize + write context (non-temporal stores;
    //         output is written once and never re-read -> keep L2 for encoder) -------
    float* obase = out + ((size_t)(b * TDLEN + row0)) * (2 * DDIM);
    const int dbase = wid * DCHUNK;
    #pragma unroll
    for (int mt = 0; mt < 2; ++mt)
        #pragma unroll
        for (int v = 0; v < 8; ++v) {
            const int r = mt * 16 + v + 8 * hi;
            const float rl = 1.0f / Lrow[r];
            #pragma unroll
            for (int nt = 0; nt < 8; ++nt)
                __builtin_nontemporal_store(
                    O[mt][nt][v] * rl,
                    obase + (size_t)r * (2 * DDIM) + dbase + nt * 16 + l16);
        }

    // ---------------- epilogue: copy decoder rows (fp32 exact, NT both ways) ----------------
    #pragma unroll
    for (int i = 0; i < (BM * DDIM) / (NTHREADS * 4); ++i) {   // 32 iters
        int idx = (tid + i * NTHREADS) * 4;
        int r = idx >> 10;
        int c = idx & (DDIM - 1);
        v4f vv = __builtin_nontemporal_load((const v4f*)(qglob + idx));
        __builtin_nontemporal_store(vv, (v4f*)(obase + (size_t)r * (2 * DDIM) + DDIM + c));
    }
}

extern "C" void kernel_launch(void* const* d_in, const int* in_sizes, int n_in,
                              void* d_out, int out_size, void* d_ws, size_t ws_size,
                              hipStream_t stream) {
    const float* enc = (const float*)d_in[0];   // [16, 2048, 1024]
    const float* dec = (const float*)d_in[1];   // [16, 1024, 1024]
    float* out = (float*)d_out;                 // [16, 1024, 2048]

    (void)in_sizes; (void)n_in; (void)out_size; (void)d_ws; (void)ws_size;

    hipFuncSetAttribute((const void*)attn_flash_kernel,
                        hipFuncAttributeMaxDynamicSharedMemorySize,
                        (int)SMEM_BYTES);

    dim3 grid(BATCH, TDLEN / BM);   // 16 x 32 workgroups
    dim3 block(NTHREADS);
    hipLaunchKernelGGL(attn_flash_kernel, grid, block, SMEM_BYTES, stream,
                       enc, dec, out);
}
